// RNN_82240033784239
// MI455X (gfx1250) — matrix operand order repaired and checked by reading, not compiled, not run on hardware
//
#include <hip/hip_runtime.h>
#include <hip/hip_bf16.h>
#include <cstdint>

// RNN: out[n,t,h] = scan over t of h = tanh(h @ Wh + (x@Wx + b)[:,t,:])
// N=64, T=512, D=1024, H=1024.
//
// Phase A: convert Wx, Wh, h0 to bf16 in workspace.
// Phase B: xW = x@Wx + b (bf16 WMMA, f32 accum), double-buffered pipeline:
//          B tiles via TDM tensor_load_to_lds (HW-padded rows), A tiles via
//          f32->bf16 convert. ws layout [t][n][h].
// Phase C: persistent 16-block scan; Wh slice TDM-preloaded into LDS (144KB),
//          h ping-pong bf16 in L2, software grid barrier, HW tanh.

#define N_B   64
#define T_S   512
#define D_IN  1024
#define H_DIM 1024

typedef __bf16        v16bf __attribute__((ext_vector_type(16)));
typedef float         v8f   __attribute__((ext_vector_type(8)));
typedef unsigned int  v4u   __attribute__((ext_vector_type(4)));
typedef unsigned int  v8u   __attribute__((ext_vector_type(8)));
typedef unsigned int  u32x4 __attribute__((ext_vector_type(4)));
typedef int           i32x8 __attribute__((ext_vector_type(8)));
typedef int           i32x4 __attribute__((ext_vector_type(4)));

#if defined(__HIP_DEVICE_COMPILE__) && __has_builtin(__builtin_amdgcn_tensor_load_to_lds)
#define USE_TDM 1
#else
#define USE_TDM 0
#endif

static __device__ __forceinline__ unsigned short f2bf(float f) {
    union { float f; unsigned u; } c; c.f = f;
    return (unsigned short)((c.u + 0x7FFFu + ((c.u >> 16) & 1u)) >> 16);
}

// Pack two f32 -> packed bf16x2 (one VALU op on CDNA5 if builtin exists).
static __device__ __forceinline__ unsigned pk_bf16(float a, float b) {
#if defined(__HIP_DEVICE_COMPILE__) && __has_builtin(__builtin_amdgcn_cvt_pk_bf16_f32)
    auto r = __builtin_amdgcn_cvt_pk_bf16_f32(a, b);
    static_assert(sizeof(r) == 4, "cvt_pk_bf16 size");
    return __builtin_bit_cast(unsigned, r);
#else
    return (unsigned)f2bf(a) | ((unsigned)f2bf(b) << 16);
#endif
}

// Hardware TANH trans op (TRANS32 on CDNA5; co-executes with XDL ops).
static __device__ __forceinline__ float fast_tanh(float x) {
#if defined(__HIP_DEVICE_COMPILE__) && __has_builtin(__builtin_amdgcn_tanhf)
    return __builtin_amdgcn_tanhf(x);
#elif defined(__HIP_DEVICE_COMPILE__) && __has_builtin(__builtin_amdgcn_tanh_f32)
    return __builtin_amdgcn_tanh_f32(x);
#else
    return tanhf(x);
#endif
}

// Load a 16-element bf16 fragment as two 16B chunks at p and p+OFS (elements).
// A-fragments: OFS=16 (K gap per ISA layout). B-fragments: gathered per element in the same k order.
template <int OFS>
static __device__ __forceinline__ v16bf load_frag(const unsigned short* p) {
    v4u lo = *(const v4u*)p;
    v4u hi = *(const v4u*)(p + OFS);
    v8u w = { lo.x, lo.y, lo.z, lo.w, hi.x, hi.y, hi.z, hi.w };
    return __builtin_bit_cast(v16bf, w);
}

static __device__ __forceinline__ v8f wmma_bf16(v16bf a, v16bf b, v8f c) {
    return __builtin_amdgcn_wmma_f32_16x16x32_bf16(
        false, a, false, b, (short)0, c, false, false);
}

#if USE_TDM
// TDM: DMA a 2D tile (tile_d0 cols x tile_d1 rows, bf16) from a row-major
// [*, H_DIM] bf16 tensor into LDS with hardware row padding.
// pad_int: pad after 2<<pad_int DWORDs; pad_amt: insert (pad_amt+1) DWORDs.
// clang-23 (amdgpu-toolchain) 6-arg form: (g0, g1, g2, g3, extra8, cpol).
static __device__ __forceinline__ void tdm_load_2d(
    const void* gtile, unsigned lds_off,
    unsigned tile_d0, unsigned tile_d1, unsigned pad_int, unsigned pad_amt)
{
    unsigned long long ga = (unsigned long long)(uintptr_t)gtile;
    u32x4 g0 = { 1u,                          // count=1, user flags 0
                 lds_off,                     // [63:32]  lds_addr
                 (unsigned)ga,                // [95:64]  global_addr lo
                 (unsigned)((ga >> 32) & 0x1FFFFFFu) | (2u << 30) }; // type=2
    i32x8 g1 = {
        (int)((1u << 16) | (1u << 20) | (pad_int << 22) | (pad_amt << 25)),
        (int)((H_DIM & 0xFFFFu) << 16),       // tensor_dim0 lo16 @ [63:48]
        (int)((H_DIM & 0xFFFFu) << 16),       // dim0 hi16=0, tensor_dim1 lo16 @ [95:80]
        (int)(tile_d0 << 16),                 // tile_dim0 @ [127:112]
        (int)tile_d1,                         // tile_dim1 @ [143:128]
        (int)H_DIM,                           // tensor_dim0_stride lo32
        0, 0 };
    i32x4 z4 = { 0, 0, 0, 0 };
    i32x8 z8 = { 0, 0, 0, 0, 0, 0, 0, 0 };
    __builtin_amdgcn_tensor_load_to_lds(g0, g1, z4, z4, z8, 0);
}
#endif

// ---------------------------------------------------------------- Phase A ---
__global__ __launch_bounds__(256) void f32_to_bf16_kernel(
    const float* __restrict__ src, unsigned short* __restrict__ dst, int n4) {
    int i = blockIdx.x * 256 + threadIdx.x;
    if (i < n4) {
        float4 v = *(const float4*)(src + (size_t)i * 4);
        uint2 p;
        p.x = pk_bf16(v.x, v.y);
        p.y = pk_bf16(v.z, v.w);
        *(uint2*)(dst + (size_t)i * 4) = p;
    }
}

// ---------------------------------------------------------------- Phase B ---
// C[32768,1024] = A(x f32->bf16) * B(Wx bf16) + bias. Block 128x128, K-tile 32,
// 8 waves (wave tile 32x64 = 2x4 WMMA). Double-buffered LDS pipeline.
#define BM 128
#define BN 128
#define BK 32
#define LDA_S 48    // elems; 96B A row stride
#define LDB_S 136   // elems; 272B B row stride (= 256B row + 16B TDM pad)

__global__ __launch_bounds__(256) void gemm_xw_kernel(
    const float* __restrict__ x,            // [N*T, D]
    const unsigned short* __restrict__ wxb, // [D, H] bf16
    const float* __restrict__ bias,         // [H]
    float* __restrict__ xws)                // [T][N][H] f32
{
    __shared__ unsigned short As[2][BM * LDA_S]; // 2 x 12 KB
    __shared__ unsigned short Bs[2][BK * LDB_S]; // 2 x 8.5 KB

    const int tid  = threadIdx.x;
    const int lane = tid & 31;
    const int wave = tid >> 5;
    const int wy   = wave & 3;
    const int wx   = wave >> 2;
    const int m_blk = blockIdx.x * BM;
    const int n_blk = blockIdx.y * BN;
    const int half = lane >> 4;
    const int l15  = lane & 15;

    v8f c[2][4];
    #pragma unroll
    for (int nt = 0; nt < 4; ++nt) {
        float bv = bias[n_blk + wx * 64 + nt * 16 + l15];
        #pragma unroll
        for (int mt = 0; mt < 2; ++mt)
            #pragma unroll
            for (int v = 0; v < 8; ++v) c[mt][nt][v] = bv;
    }

    // ---- staging helpers -------------------------------------------------
    auto stage_a = [&](int kt, int buf) {
        #pragma unroll
        for (int j = 0; j < 4; ++j) {
            int f   = tid + 256 * j;
            int row = f >> 3;
            int ch  = f & 7;
            float4 v4 = *(const float4*)(x + (size_t)(m_blk + row) * D_IN + kt + ch * 4);
            uint2 p;
            p.x = pk_bf16(v4.x, v4.y);
            p.y = pk_bf16(v4.z, v4.w);
            *(uint2*)(&As[buf][row * LDA_S + ch * 4]) = p;
        }
    };
    auto stage_b = [&](int kt, int buf) {
#if USE_TDM
        if (wave == 0) {
            // 128 cols x 32 rows bf16; row=256B=64 DW -> pad_int=5; pad 4 DW -> pad_amt=3.
            tdm_load_2d(wxb + (size_t)kt * H_DIM + n_blk,
                        (unsigned)(uintptr_t)(void*)&Bs[buf][0],
                        /*tile_d0=*/BN, /*tile_d1=*/BK, /*pad_int=*/5, /*pad_amt=*/3);
        }
#else
        #pragma unroll
        for (int j = 0; j < 2; ++j) {
            int cidx = tid + 256 * j;
            int krow = cidx >> 4;
            int ch   = cidx & 15;
            v4u bv = *(const v4u*)(wxb + (size_t)(kt + krow) * H_DIM + n_blk + ch * 8);
            *(v4u*)(&Bs[buf][krow * LDB_S + ch * 8]) = bv;
        }
#endif
    };

    // ---- prologue --------------------------------------------------------
    stage_b(0, 0);
    stage_a(0, 0);
#if USE_TDM
    if (wave == 0) __builtin_amdgcn_s_wait_tensorcnt(0);
#endif
    __syncthreads();

    // ---- pipelined main loop --------------------------------------------
    for (int kt = 0; kt < D_IN; kt += BK) {
        const int cur = (kt >> 5) & 1;
        const int nxt = cur ^ 1;
        const bool more = (kt + BK) < D_IN;
        if (more) {
            stage_b(kt + BK, nxt);   // async DMA in flight during compute
            stage_a(kt + BK, nxt);
        }

        v16bf af[2];
        #pragma unroll
        for (int mt = 0; mt < 2; ++mt) {
            int row = wy * 32 + mt * 16 + l15;
            af[mt] = load_frag<16>(&As[cur][row * LDA_S + half * 8]);
        }
        v16bf bfr[4];
        {
            #pragma unroll
            for (int nt = 0; nt < 4; ++nt)
                #pragma unroll
                for (int s = 0; s < 16; ++s)
                    bfr[nt][s] = __builtin_bit_cast(__bf16, Bs[cur][((s & 7) + 8 * half + 16 * (s >> 3)) * LDB_S + wx * 64 + nt * 16 + l15]);
        }
        #pragma unroll
        for (int mt = 0; mt < 2; ++mt)
            #pragma unroll
            for (int nt = 0; nt < 4; ++nt)
                c[mt][nt] = wmma_bf16(af[mt], bfr[nt], c[mt][nt]);

#if USE_TDM
        if (wave == 0 && more) __builtin_amdgcn_s_wait_tensorcnt(0);
#endif
        __syncthreads();
    }

    // ---- epilogue: store to xws[t][n][h]; global row r = n*T + t ---------
    #pragma unroll
    for (int mt = 0; mt < 2; ++mt) {
        int row = m_blk + wy * 32 + mt * 16 + half * 8;
        #pragma unroll
        for (int nt = 0; nt < 4; ++nt) {
            int col = n_blk + wx * 64 + nt * 16 + l15;
            #pragma unroll
            for (int v = 0; v < 8; ++v) {
                int r  = row + v;
                int t  = r & (T_S - 1);
                int nb = r >> 9;
                xws[(size_t)t * (N_B * H_DIM) + (size_t)nb * H_DIM + col] = c[mt][nt][v];
            }
        }
    }
}

// ---------------------------------------------------------------- Phase C ---
// Persistent scan. 16 blocks x 256 threads; block owns 64 output columns.
// Wh slice [1024 x 64] bf16 resident in LDS, stride 72 elems (144B) -> 144 KB.
#define P2_BLOCKS 16
#define LDW_S 72

__global__ __launch_bounds__(256) void rnn_scan_kernel(
    const unsigned short* __restrict__ whb, // [H][H] bf16
    const float* __restrict__ xws,          // [T][N][H]
    unsigned short* __restrict__ hb,        // [2][N][H] bf16 ping-pong
    float* __restrict__ out,                // [N][T][H]
    unsigned int* __restrict__ bar)         // bar[0]=count, bar[32]=epoch
{
    __shared__ unsigned short Bs[H_DIM * LDW_S]; // 144 KB

    const int tid   = threadIdx.x;
    const int lane  = tid & 31;
    const int wave  = tid >> 5;
    const int wy    = wave & 3;
    const int wx    = wave >> 2;
    const int n_blk = blockIdx.x * 64;
    const int half  = lane >> 4;
    const int l15   = lane & 15;
    const int m0    = wy * 16;

    // One-time Wh slice preload: 64 cols x 1024 rows bf16.
    // Row = 128B = 32 DW -> pad_int=4; pad 4 DW (16B) -> pad_amt=3.
#if USE_TDM
    if (wave == 0) {
        tdm_load_2d(whb + n_blk, (unsigned)(uintptr_t)(void*)&Bs[0],
                    /*tile_d0=*/64, /*tile_d1=*/H_DIM, /*pad_int=*/4, /*pad_amt=*/3);
        __builtin_amdgcn_s_wait_tensorcnt(0);
    }
#else
    for (int i = tid; i < H_DIM * 4; i += 256) {
        int k  = i >> 2;
        int ch = i & 3;
        v4u v = *(const v4u*)(whb + (size_t)k * H_DIM + n_blk + ch * 8);
        *(v4u*)(&Bs[k * LDW_S + ch * 8]) = v;
    }
#endif
    __syncthreads();

    unsigned epoch_local = 0;

    for (int t = 0; t < T_S; ++t) {
        const unsigned short* hprev = hb + (size_t)(t & 1) * (N_B * H_DIM);
        unsigned short*       hnext = hb + (size_t)((t + 1) & 1) * (N_B * H_DIM);

        v8f c[2];
        #pragma unroll
        for (int j = 0; j < 2; ++j) {
            int n = n_blk + wx * 32 + j * 16 + l15;
            const float* xp = xws + (size_t)t * (N_B * H_DIM)
                                  + (size_t)(m0 + half * 8) * H_DIM + n;
            #pragma unroll
            for (int v = 0; v < 8; ++v) c[j][v] = xp[(size_t)v * H_DIM];
        }

        #pragma unroll 4
        for (int kt = 0; kt < H_DIM; kt += 32) {
            int arow = m0 + l15;
            v16bf a = load_frag<16>(hprev + (size_t)arow * H_DIM + kt + half * 8);
            v16bf b0, b1;
            #pragma unroll
            for (int s = 0; s < 16; ++s) {
                int krow = kt + (s & 7) + 8 * half + 16 * (s >> 3);
                b0[s] = __builtin_bit_cast(__bf16, Bs[krow * LDW_S + wx * 32 + l15]);
                b1[s] = __builtin_bit_cast(__bf16, Bs[krow * LDW_S + wx * 32 + 16 + l15]);
            }
            c[0] = wmma_bf16(a, b0, c[0]);
            c[1] = wmma_bf16(a, b1, c[1]);
        }

        #pragma unroll
        for (int j = 0; j < 2; ++j) {
            int n = n_blk + wx * 32 + j * 16 + l15;
            #pragma unroll
            for (int v = 0; v < 8; ++v) {
                int m = m0 + half * 8 + v;
                float hv = fast_tanh(c[j][v]);
                out[(size_t)m * (T_S * H_DIM) + (size_t)t * H_DIM + n] = hv;
                hnext[(size_t)m * H_DIM + n] = f2bf(hv);
            }
        }

        __threadfence();
        __syncthreads();
        if (tid == 0) {
            unsigned prev = __hip_atomic_fetch_add(&bar[0], 1u, __ATOMIC_ACQ_REL,
                                                   __HIP_MEMORY_SCOPE_AGENT);
            if (prev == P2_BLOCKS - 1) {
                __hip_atomic_store(&bar[0], 0u, __ATOMIC_RELAXED,
                                   __HIP_MEMORY_SCOPE_AGENT);
                __hip_atomic_fetch_add(&bar[32], 1u, __ATOMIC_RELEASE,
                                       __HIP_MEMORY_SCOPE_AGENT);
            } else {
                while (__hip_atomic_load(&bar[32], __ATOMIC_ACQUIRE,
                                         __HIP_MEMORY_SCOPE_AGENT) <= epoch_local) {
                    __builtin_amdgcn_s_sleep(8);
                }
            }
        }
        ++epoch_local;
        __syncthreads();
    }
}

// ---------------------------------------------------------------- launch ---
extern "C" void kernel_launch(void* const* d_in, const int* in_sizes, int n_in,
                              void* d_out, int out_size, void* d_ws, size_t ws_size,
                              hipStream_t stream) {
    (void)in_sizes; (void)n_in; (void)out_size; (void)ws_size;
    const float* x  = (const float*)d_in[0];
    const float* h0 = (const float*)d_in[1];
    const float* Wx = (const float*)d_in[2];
    const float* Wh = (const float*)d_in[3];
    const float* b  = (const float*)d_in[4];
    float* out = (float*)d_out;

    char* ws = (char*)d_ws;
    float*          xws = (float*)ws;                                  // 128 MB
    unsigned short* wxb = (unsigned short*)(ws + (size_t)134217728);   // 2 MB
    unsigned short* whb = wxb + (size_t)D_IN * H_DIM;                  // 2 MB
    unsigned short* hb  = whb + (size_t)H_DIM * H_DIM;                 // 256 KB
    unsigned int*   bar = (unsigned int*)(hb + (size_t)2 * N_B * H_DIM);

    (void)hipMemsetAsync(bar, 0, 256, stream);
    f32_to_bf16_kernel<<<(D_IN * H_DIM / 4 + 255) / 256, 256, 0, stream>>>(Wx, wxb, D_IN * H_DIM / 4);
    f32_to_bf16_kernel<<<(H_DIM * H_DIM / 4 + 255) / 256, 256, 0, stream>>>(Wh, whb, H_DIM * H_DIM / 4);
    f32_to_bf16_kernel<<<(N_B * H_DIM / 4 + 255) / 256, 256, 0, stream>>>(h0, hb, N_B * H_DIM / 4);

    dim3 g1((N_B * T_S) / BM, H_DIM / BN);
    gemm_xw_kernel<<<g1, 256, 0, stream>>>(x, wxb, b, xws);

    rnn_scan_kernel<<<P2_BLOCKS, 256, 0, stream>>>(whb, xws, hb, out, bar);
}
